// EdgeMask_25159918420540
// MI455X (gfx1250) — matrix operand order repaired and checked by reading, not compiled, not run on hardware
//
#include <hip/hip_runtime.h>
#include <hip/hip_bf16.h>
#include <math.h>

#define N_NODES 50000
#define NEDGE   800000
#define D       64
#define H       128

typedef __attribute__((ext_vector_type(16))) __bf16 v16bf;
typedef __attribute__((ext_vector_type(2)))  __bf16 v2bf;
typedef __attribute__((ext_vector_type(8)))  float  v8f;
typedef __attribute__((ext_vector_type(4)))  unsigned int u32x4;
typedef __attribute__((ext_vector_type(4)))  int i32x4;
typedef __attribute__((ext_vector_type(8)))  int i32x8;

#if __has_builtin(__builtin_amdgcn_tensor_load_to_lds)
#define USE_TDM 1
#endif
#if __has_builtin(__builtin_amdgcn_cvt_pk_bf16_f32)
#define USE_CVT_PK 1
#endif

// ---------- bf16 helpers ----------
// pack two floats -> packed bf16x2 (lo = bf16(a), hi = bf16(b))
__device__ __forceinline__ unsigned pack_bf16(float a, float b) {
#ifdef USE_CVT_PK
    union { v2bf v; unsigned u; } cv;
    cv.v = __builtin_amdgcn_cvt_pk_bf16_f32(a, b);
    return cv.u;
#elif __has_builtin(__builtin_amdgcn_perm)
    // v_perm_b32: pool = {S0 bytes 4..7, S1 bytes 0..3};
    // out = {b.byte3, b.byte2, a.byte3, a.byte2}  (one VALU op, truncation)
    return __builtin_amdgcn_perm(__float_as_uint(b), __float_as_uint(a), 0x07060302u);
#else
    return (__float_as_uint(a) >> 16) | (__float_as_uint(b) & 0xFFFF0000u);
#endif
}
__device__ __forceinline__ unsigned short f2bf1(float a) {
    return (unsigned short)(__float_as_uint(a) >> 16);   // truncate
}
__device__ __forceinline__ float bf2f(unsigned short h) {
    return __uint_as_float(((unsigned int)h) << 16);
}

union FragU { v16bf bf; uint4 q[2]; };

__device__ __forceinline__ v16bf load_frag2(const unsigned short* p0,
                                            const unsigned short* p1) {
    FragU u;
    u.q[0] = *(const uint4*)p0;
    u.q[1] = *(const uint4*)p1;
    return u.bf;
}

// ---------- Tensor Data Mover: 1-D DMA global -> LDS ----------
#ifdef USE_TDM
__device__ __forceinline__ void tdm_load_1d(const void* gptr, void* lptr, unsigned nbytes) {
    unsigned long long ga = (unsigned long long)gptr;
    unsigned lds = (unsigned)(unsigned long long)
        (__attribute__((address_space(3))) char*)lptr;
    unsigned n8 = nbytes >> 3;   // 8-byte elements

    union { u32x4 v; unsigned a[4]; } g0;
    g0.a[0] = 1u;                                   // count=1, user descriptor
    g0.a[1] = lds;                                  // lds_addr
    g0.a[2] = (unsigned)ga;                         // global_addr[31:0]
    g0.a[3] = (unsigned)(ga >> 32) | (2u << 30);    // global_addr[56:32] | type=2

    union { i32x8 v; unsigned a[8]; } g1;
    #pragma unroll
    for (int i = 0; i < 8; ++i) g1.a[i] = 0u;
    g1.a[0] = (3u << 16);                 // data_size = 8 bytes
    g1.a[1] = (n8 & 0xFFFFu) << 16;       // tensor_dim0[15:0]
    g1.a[2] = (n8 >> 16) & 0xFFFFu;       // tensor_dim0[31:16]
    g1.a[3] = (n8 & 0xFFFFu) << 16;       // tile_dim0
    g1.a[4] = 0u;                         // tile_dim1/2 = 0 (1-D)
    g1.a[5] = n8;                         // tensor_dim0_stride[31:0]
    g1.a[6] = (n8 >> 16) & 0xFFFFu;       // tensor_dim0_stride[47:32]

    i32x4 z4 = {};
#if __has_include(<hip/amd_detail/amd_gfx1250_TDM.h>)
    i32x8 z8 = {};
    __builtin_amdgcn_tensor_load_to_lds(g0.v, g1.v, z4, z4, z8, 0);
#else
    __builtin_amdgcn_tensor_load_to_lds(g0.v, g1.v, z4, z4, 0);
#endif
}
#endif

// ---------------- degree histogram ----------------
__global__ void zero_i32(int* p, int n) {
    int i = blockIdx.x * blockDim.x + threadIdx.x;
    if (i < n) p[i] = 0;
}

__global__ void count_deg(const int* __restrict__ src, const int* __restrict__ dst,
                          int* __restrict__ dout, int* __restrict__ din, int n) {
    int i = blockIdx.x * blockDim.x + threadIdx.x;
    if (i < n) {
        atomicAdd(&dout[src[i]], 1);
        atomicAdd(&din[dst[i]], 1);
    }
}

// ---------------- weight transpose + bf16 convert ----------------
__global__ void conv_w(const float* __restrict__ W1, const float* __restrict__ W2,
                       unsigned short* __restrict__ w1t, unsigned short* __restrict__ w2t) {
    int i = blockIdx.x * blockDim.x + threadIdx.x;
    if (i < H * H) {
        int n = i / H, k = i % H;
        w1t[n * H + k] = f2bf1(W1[k * H + n]);
    } else {
        int j = i - H * H;
        if (j < (H / 2) * H) {
            int n = j / H, k = j % H;
            w2t[n * H + k] = f2bf1(W2[k * (H / 2) + n]);
        }
    }
}

// ---------------- fused edge MLP (WMMA bf16) ----------------
__global__ __launch_bounds__(256) void edge_mlp(
    const float* __restrict__ node_emb,
    const int* __restrict__ src, const int* __restrict__ dst,
    const float* __restrict__ noise,
    const float* __restrict__ b1v, const float* __restrict__ g1, const float* __restrict__ bt1,
    const float* __restrict__ b2v, const float* __restrict__ g2, const float* __restrict__ bt2,
    const float* __restrict__ W3, const float* __restrict__ b3,
    const unsigned short* __restrict__ w1t_g, const unsigned short* __restrict__ w2t_g,
    const int* __restrict__ deg_out, const int* __restrict__ deg_in,
    float* __restrict__ out)
{
    __shared__ __align__(16) unsigned short lds_w1[H * H];          // 32 KB, W1^T [n][k]
    __shared__ __align__(16) unsigned short lds_w2[(H / 2) * H];    // 16 KB, W2^T [n][k]
    __shared__ __align__(16) unsigned short lds_feat[8][16 * H];    // 32 KB, per-wave staging

    const int tid  = threadIdx.x;
    const int wid  = tid >> 5;
    const int lane = tid & 31;
    const int m    = lane & 15;
    const int hi   = lane >> 4;

    // ---- stage weights to LDS via Tensor Data Mover ----
#ifdef USE_TDM
    if (wid == 0) {
        tdm_load_1d(w1t_g, lds_w1, H * H * 2);
        tdm_load_1d(w2t_g, lds_w2, (H / 2) * H * 2);
        __builtin_amdgcn_s_wait_tensorcnt(0);
    }
#else
    {
        const uint4* gw1 = (const uint4*)w1t_g;
        uint4* lw1 = (uint4*)lds_w1;
        for (int i = tid; i < (H * H) / 8; i += 256) lw1[i] = gw1[i];
        const uint4* gw2 = (const uint4*)w2t_g;
        uint4* lw2 = (uint4*)lds_w2;
        for (int i = tid; i < ((H / 2) * H) / 8; i += 256) lw2[i] = gw2[i];
    }
#endif
    __syncthreads();

    unsigned short* feat = lds_feat[wid];
    const int base = (blockIdx.x * 8 + wid) * 16;
    const int e  = base + m;
    const int ec = (e < NEDGE) ? e : (NEDGE - 1);

    // ---- gather: lane (m,hi) fills half of edge m's concat feature row ----
    {
        const int node = (hi == 0) ? src[ec] : dst[ec];
        const float4* np = (const float4*)(node_emb + (long)node * D);
        unsigned short* fp = feat + m * H + hi * D;
        #pragma unroll
        for (int c = 0; c < D / 4; ++c) {
            float4 v = np[c];
            uint2 pk;
            pk.x = pack_bf16(v.x, v.y);
            pk.y = pack_bf16(v.z, v.w);
            *(uint2*)(fp + c * 4) = pk;
        }
    }

    // ---- layer 1: [16x128] @ [128x128] via 8 N-tiles x 4 K-blocks of WMMA ----
    v8f acc[8];
    #pragma unroll
    for (int nt = 0; nt < 8; ++nt) acc[nt] = (v8f){};

    v16bf afrag[4];
    #pragma unroll
    for (int kb = 0; kb < 4; ++kb) {
        const unsigned short* p0 = feat + m * H + kb * 32 + hi * 8;
        afrag[kb] = load_frag2(p0, p0 + 16);
    }
    #pragma unroll
    for (int nt = 0; nt < 8; ++nt) {
        #pragma unroll
        for (int kb = 0; kb < 4; ++kb) {
            const unsigned short* p0 = lds_w1 + (nt * 16 + m) * H + kb * 32 + hi * 8;
            v16bf bfrag = load_frag2(p0, p0 + 16);
            acc[nt] = __builtin_amdgcn_wmma_f32_16x16x32_bf16(
                false, afrag[kb], false, bfrag, (short)0, acc[nt], false, false);
        }
    }

    // ---- write h1 (+bias) back to LDS as bf16 rows [m][k] ----
    #pragma unroll
    for (int nt = 0; nt < 8; ++nt) {
        float bias = b1v[nt * 16 + m];
        #pragma unroll
        for (int v = 0; v < 8; ++v) {
            feat[(v + 8 * hi) * H + nt * 16 + m] = f2bf1(acc[nt][v] + bias);
        }
    }

    // ---- LayerNorm 1 + ReLU (lane pair per edge row, shfl_xor(16) reduce) ----
    {
        unsigned short* row = feat + m * H + hi * 64;
        uint4 qs[8];
        float s = 0.f, ss = 0.f;
        #pragma unroll
        for (int c = 0; c < 8; ++c) {
            qs[c] = *(const uint4*)(row + c * 8);
            unsigned w[4] = {qs[c].x, qs[c].y, qs[c].z, qs[c].w};
            #pragma unroll
            for (int t = 0; t < 4; ++t) {
                float x0 = bf2f((unsigned short)w[t]);
                float x1 = bf2f((unsigned short)(w[t] >> 16));
                s += x0 + x1; ss += x0 * x0 + x1 * x1;
            }
        }
        s  += __shfl_xor(s, 16, 32);
        ss += __shfl_xor(ss, 16, 32);
        float mean = s * (1.0f / 128.0f);
        float var  = ss * (1.0f / 128.0f) - mean * mean;
        float rstd = rsqrtf(var + 1e-5f);
        #pragma unroll
        for (int c = 0; c < 8; ++c) {
            unsigned w[4] = {qs[c].x, qs[c].y, qs[c].z, qs[c].w};
            unsigned ow[4];
            #pragma unroll
            for (int t = 0; t < 4; ++t) {
                int k = hi * 64 + c * 8 + t * 2;
                float x0 = bf2f((unsigned short)w[t]);
                float x1 = bf2f((unsigned short)(w[t] >> 16));
                float y0 = (x0 - mean) * rstd * g1[k]     + bt1[k];
                float y1 = (x1 - mean) * rstd * g1[k + 1] + bt1[k + 1];
                y0 = y0 > 0.f ? y0 : 0.f;
                y1 = y1 > 0.f ? y1 : 0.f;
                ow[t] = pack_bf16(y0, y1);
            }
            uint4 o; o.x = ow[0]; o.y = ow[1]; o.z = ow[2]; o.w = ow[3];
            *(uint4*)(row + c * 8) = o;
        }
    }

    // ---- layer 2: [16x128] @ [128x64] via 4 N-tiles x 4 K-blocks ----
    v8f acc2[4];
    #pragma unroll
    for (int nt = 0; nt < 4; ++nt) acc2[nt] = (v8f){};

    v16bf a2[4];
    #pragma unroll
    for (int kb = 0; kb < 4; ++kb) {
        const unsigned short* p0 = feat + m * H + kb * 32 + hi * 8;
        a2[kb] = load_frag2(p0, p0 + 16);
    }
    #pragma unroll
    for (int nt = 0; nt < 4; ++nt) {
        #pragma unroll
        for (int kb = 0; kb < 4; ++kb) {
            const unsigned short* p0 = lds_w2 + (nt * 16 + m) * H + kb * 32 + hi * 8;
            v16bf bfrag = load_frag2(p0, p0 + 16);
            acc2[nt] = __builtin_amdgcn_wmma_f32_16x16x32_bf16(
                false, a2[kb], false, bfrag, (short)0, acc2[nt], false, false);
        }
    }

    // ---- write h2 (+bias) as bf16 rows with stride 64 (reuse staging) ----
    #pragma unroll
    for (int nt = 0; nt < 4; ++nt) {
        float bias = b2v[nt * 16 + m];
        #pragma unroll
        for (int v = 0; v < 8; ++v) {
            feat[(v + 8 * hi) * 64 + nt * 16 + m] = f2bf1(acc2[nt][v] + bias);
        }
    }

    // ---- LayerNorm 2 + ReLU + dot W3 + gate + degree norm ----
    {
        const unsigned short* row = feat + m * 64 + hi * 32;
        uint4 qs[4];
        float s = 0.f, ss = 0.f;
        #pragma unroll
        for (int c = 0; c < 4; ++c) {
            qs[c] = *(const uint4*)(row + c * 8);
            unsigned w[4] = {qs[c].x, qs[c].y, qs[c].z, qs[c].w};
            #pragma unroll
            for (int t = 0; t < 4; ++t) {
                float x0 = bf2f((unsigned short)w[t]);
                float x1 = bf2f((unsigned short)(w[t] >> 16));
                s += x0 + x1; ss += x0 * x0 + x1 * x1;
            }
        }
        s  += __shfl_xor(s, 16, 32);
        ss += __shfl_xor(ss, 16, 32);
        float mean = s * (1.0f / 64.0f);
        float var  = ss * (1.0f / 64.0f) - mean * mean;
        float rstd = rsqrtf(var + 1e-5f);

        float dot = 0.f;
        #pragma unroll
        for (int c = 0; c < 4; ++c) {
            unsigned w[4] = {qs[c].x, qs[c].y, qs[c].z, qs[c].w};
            #pragma unroll
            for (int t = 0; t < 4; ++t) {
                int k = hi * 32 + c * 8 + t * 2;
                float x0 = bf2f((unsigned short)w[t]);
                float x1 = bf2f((unsigned short)(w[t] >> 16));
                float y0 = (x0 - mean) * rstd * g2[k]     + bt2[k];
                float y1 = (x1 - mean) * rstd * g2[k + 1] + bt2[k + 1];
                y0 = y0 > 0.f ? y0 : 0.f;
                y1 = y1 > 0.f ? y1 : 0.f;
                dot += y0 * W3[k] + y1 * W3[k + 1];
            }
        }
        dot += __shfl_xor(dot, 16, 32);

        if (hi == 0 && e < NEDGE) {
            float u = noise[e];
            float z = logf(u) - log1pf(-u) + dot + b3[0];
            float gate = 1.0f / (1.0f + expf(-z));
            int od = deg_out[src[e]]; od = od < 1 ? 1 : od;
            int id = deg_in[dst[e]];  id = id < 1 ? 1 : id;
            out[e] = gate * rsqrtf((float)od) * rsqrtf((float)id);
        }
    }
}

extern "C" void kernel_launch(void* const* d_in, const int* in_sizes, int n_in,
                              void* d_out, int out_size, void* d_ws, size_t ws_size,
                              hipStream_t stream) {
    const float* node_emb = (const float*)d_in[0];
    const int*   src      = (const int*)d_in[1];
    const int*   dst      = (const int*)d_in[2];
    const float* noise    = (const float*)d_in[3];
    const float* W1       = (const float*)d_in[4];
    const float* b1       = (const float*)d_in[5];
    const float* g1       = (const float*)d_in[6];
    const float* bt1      = (const float*)d_in[7];
    const float* W2       = (const float*)d_in[8];
    const float* b2       = (const float*)d_in[9];
    const float* g2       = (const float*)d_in[10];
    const float* bt2      = (const float*)d_in[11];
    const float* W3       = (const float*)d_in[12];
    const float* b3       = (const float*)d_in[13];

    char* ws = (char*)d_ws;
    int* deg_out = (int*)ws;                              // 200000 B
    int* deg_in  = (int*)(ws + 200000);                   // 200000 B (contiguous)
    unsigned short* w1t = (unsigned short*)(ws + 400000); // 32768 B
    unsigned short* w2t = (unsigned short*)(ws + 432768); // 16384 B

    zero_i32<<<(2 * N_NODES + 255) / 256, 256, 0, stream>>>(deg_out, 2 * N_NODES);
    count_deg<<<(NEDGE + 255) / 256, 256, 0, stream>>>(src, dst, deg_out, deg_in, NEDGE);
    conv_w<<<((H * H + (H / 2) * H) + 255) / 256, 256, 0, stream>>>(W1, W2, w1t, w2t);

    edge_mlp<<<(NEDGE + 127) / 128, 256, 0, stream>>>(
        node_emb, src, dst, noise,
        b1, g1, bt1, b2, g2, bt2, W3, b3,
        w1t, w2t, deg_out, deg_in, (float*)d_out);
}
